// Sam3DetrDecoder_59176059404291
// MI455X (gfx1250) — matrix operand order repaired and checked by reading, not compiled, not run on hardware
//
#include <hip/hip_runtime.h>
#include <hip/hip_bf16.h>

// ---------------------------------------------------------------------------
// SAM3 DETR decoder for MI455X (gfx1250, wave32, WMMA).
// All GEMMs + attention products use v_wmma_f32_16x16x32_f16 (f16 in, f32 acc).
// GEMM: 128x64 block tile, 8 waves each computing 32x32 (4 WMMAs/k-step,
// fragments reused 2x). Fragment loads are b128-vectorized.
// ---------------------------------------------------------------------------

typedef __attribute__((ext_vector_type(16))) _Float16 v16h;
typedef __attribute__((ext_vector_type(8)))  _Float16 v8h;
typedef __attribute__((ext_vector_type(8)))  float    v8f;
typedef __attribute__((ext_vector_type(4)))  float    v4f;

#define GF_RELU  1
#define GF_RESID 2

#define BIAS_NONE 0
#define BIAS_TEXT 1
#define BIAS_VIS  2

static __device__ __forceinline__ v8f wmma_f16(v16h a, v16h b, v8f c) {
  return __builtin_amdgcn_wmma_f32_16x16x32_f16(false, a, false, b, (short)0, c,
                                                false, false);
}

static __device__ __forceinline__ v16h comb(v8h lo, v8h hi) {
  v16h r;
#pragma unroll
  for (int i = 0; i < 8; ++i) { r[i] = lo[i]; r[8 + i] = hi[i]; }
  return r;
}

// 16 consecutive f32 -> v16h
static __device__ __forceinline__ v16h ld16h(const float* p) {
  v16h r;
  const v4f* p4 = (const v4f*)p;
#pragma unroll
  for (int i = 0; i < 4; ++i) {
    v4f t = p4[i];
#pragma unroll
    for (int j = 0; j < 4; ++j) r[i * 4 + j] = (_Float16)t[j];
  }
  return r;
}

// ---------------------------------------------------------------------------
// Generic GEMM: C(MxN) = A(MxK,f32) * W(KxN,f32) [+bias] [+resid] [relu]
// Requires K % 32 == 0 (true for all calls: 256/512/2048).
// ---------------------------------------------------------------------------
__global__ __launch_bounds__(256) void k_gemm(
    const float* __restrict__ A, const float* __restrict__ W,
    const float* __restrict__ bias, const float* __restrict__ resid,
    float* __restrict__ C, int M, int N, int K, int flags) {
  __shared__ _Float16 As[128][40];   // A tile, row-major (row, k)
  __shared__ _Float16 Bst[64][40];   // W tile, transposed  (n, k)
  const int tid = threadIdx.x;
  const int lane = tid & 31, wv = tid >> 5;
  const int g = lane >> 4, ln = lane & 15;
  const int m0 = blockIdx.x * 128, n0 = blockIdx.y * 64;
  const int wm = (wv & 3) * 32, wn = (wv >> 2) * 32;
  const bool nfull = (n0 + 64 <= N);
  v8f acc00 = {}, acc01 = {}, acc10 = {}, acc11 = {};

  const int ar = tid >> 1, ac0 = (tid & 1) * 16;  // A stage: row, col0 (16 f32)
  const int gr = m0 + ar;
  const bool arok = gr < M;
  const float* aprow = A + (size_t)(arok ? gr : (M - 1)) * K + ac0;
  const int br = tid >> 3, bc0 = (tid & 7) * 8;   // B stage: k-row, n-col0

  for (int k0 = 0; k0 < K; k0 += 32) {
    {  // stage A 128x32 (f32 -> f16), unconditional vector loads + select
      const v4f* a4 = (const v4f*)(aprow + k0);
      v4f t0 = a4[0], t1 = a4[1], t2 = a4[2], t3 = a4[3];
      v8h s0, s1;
#pragma unroll
      for (int i = 0; i < 4; ++i) {
        s0[i] = (_Float16)(arok ? t0[i] : 0.f);
        s0[4 + i] = (_Float16)(arok ? t1[i] : 0.f);
        s1[i] = (_Float16)(arok ? t2[i] : 0.f);
        s1[4 + i] = (_Float16)(arok ? t3[i] : 0.f);
      }
      *(v8h*)&As[ar][ac0] = s0;
      *(v8h*)&As[ar][ac0 + 8] = s1;
      if (arok && k0 + 32 < K) __builtin_prefetch(aprow + k0 + 32, 0, 3);
    }
    {  // stage W 32x64 transposed into Bst[n][k]
      const float* wp = W + (size_t)(k0 + br) * N;
      _Float16 hv[8];
      if (nfull) {
        const v4f* w4 = (const v4f*)(wp + n0 + bc0);
        v4f t0 = w4[0], t1 = w4[1];
#pragma unroll
        for (int i = 0; i < 4; ++i) {
          hv[i] = (_Float16)t0[i];
          hv[4 + i] = (_Float16)t1[i];
        }
      } else {
#pragma unroll
        for (int i = 0; i < 8; ++i) {
          int gc = n0 + bc0 + i;
          float v = wp[(gc < N) ? gc : (N - 1)];
          hv[i] = (_Float16)((gc < N) ? v : 0.f);
        }
      }
#pragma unroll
      for (int i = 0; i < 8; ++i) Bst[bc0 + i][br] = hv[i];
    }
    __syncthreads();
    const _Float16* a0r = &As[wm + ln][0];
    const _Float16* a1r = &As[wm + 16 + ln][0];
    const _Float16* b0r = &Bst[wn + ln][0];
    const _Float16* b1r = &Bst[wn + 16 + ln][0];
    v16h af0 = comb(*(const v8h*)(a0r + g * 8), *(const v8h*)(a0r + 16 + g * 8));
    v16h af1 = comb(*(const v8h*)(a1r + g * 8), *(const v8h*)(a1r + 16 + g * 8));
    v16h bf0 = comb(*(const v8h*)(b0r + g * 8), *(const v8h*)(b0r + 16 + g * 8));
    v16h bf1 = comb(*(const v8h*)(b1r + g * 8), *(const v8h*)(b1r + 16 + g * 8));
    acc00 = wmma_f16(af0, bf0, acc00);
    acc01 = wmma_f16(af0, bf1, acc01);
    acc10 = wmma_f16(af1, bf0, acc10);
    acc11 = wmma_f16(af1, bf1, acc11);
    __syncthreads();
  }
#pragma unroll
  for (int mi = 0; mi < 2; ++mi) {
#pragma unroll
    for (int r = 0; r < 8; ++r) {
      int row = m0 + wm + mi * 16 + g * 8 + r;
      if (row >= M) continue;
      int c0g = n0 + wn + ln, c1g = c0g + 16;
      float v0 = mi ? acc10[r] : acc00[r];
      float v1 = mi ? acc11[r] : acc01[r];
      if (c0g < N) {
        if (bias) v0 += bias[c0g];
        if (flags & GF_RESID) v0 += resid[(size_t)row * N + c0g];
        if (flags & GF_RELU) v0 = fmaxf(v0, 0.f);
        C[(size_t)row * N + c0g] = v0;
      }
      if (c1g < N) {
        if (bias) v1 += bias[c1g];
        if (flags & GF_RESID) v1 += resid[(size_t)row * N + c1g];
        if (flags & GF_RELU) v1 = fmaxf(v1, 0.f);
        C[(size_t)row * N + c1g] = v1;
      }
    }
  }
}

// ---------------------------------------------------------------------------
// Flash attention, head_dim=32, D=256, NH=8. Block = (qtile, head, batch),
// 4 waves split keys (chunks of 32), online softmax, LDS merge at end.
// ---------------------------------------------------------------------------
__global__ __launch_bounds__(128) void k_flash(
    const float* __restrict__ Q, const float* __restrict__ Kb,
    const float* __restrict__ Vb, float* __restrict__ O,
    const float* __restrict__ dy_e, const float* __restrict__ dx_e,
    const bool* __restrict__ tmask, int Lq, int Lk, int Hh, int Wd,
    int bias_mode) {
  const int Dm = 256, HD = 32;
  const int qt = blockIdx.x, h = blockIdx.y, b = blockIdx.z;
  const int tid = threadIdx.x, lane = tid & 31, wv = tid >> 5;
  const int g = lane >> 4, ln = lane & 15;
  __shared__ _Float16 Ps[4][16 * 32];
  __shared__ _Float16 Vst[4][32][40];  // V chunk transposed: [wave][dim][key]
  __shared__ float Ms[4][16], Ls[4][16], Os[4][16 * 32];
  const float scale = 0.17677669529663687f;  // 1/sqrt(32)

  // Q fragment (A layout: 16 contiguous dims at g*16), pre-scaled, OOB rows = 0
  v16h qa;
  {
    int q = qt * 16 + ln;
    float sc = (q < Lq) ? scale : 0.f;
    const float* qp = Q + ((size_t)b * Lq + ((q < Lq) ? q : (Lq - 1))) * Dm + h * HD;
    const v4f* qa4 = (const v4f*)(qp + g * 16);
    const v4f* qb4 = (const v4f*)(qp + g * 16 + 8);
    v4f t0 = qa4[0], t1 = qa4[1], t2 = qb4[0], t3 = qb4[1];
#pragma unroll
    for (int i = 0; i < 4; ++i) {
      qa[i] = (_Float16)(t0[i] * sc);
      qa[4 + i] = (_Float16)(t1[i] * sc);
      qa[8 + i] = (_Float16)(t2[i] * sc);
      qa[12 + i] = (_Float16)(t3[i] * sc);
    }
  }
  float rm[8], rl[8];
#pragma unroll
  for (int r = 0; r < 8; ++r) { rm[r] = -1e30f; rl[r] = 0.f; }
  v8f o0 = {}, o1 = {};

  for (int c = wv; c * 32 < Lk; c += 4) {
    int k0 = c * 32;
    // stage V chunk transposed (lane -> key row, 8 x b128 loads, b16 scatter)
    {
      int vkey = k0 + lane;
      const float* vp =
          Vb + ((size_t)b * Lk + ((vkey < Lk) ? vkey : (Lk - 1))) * Dm + h * HD;
      const v4f* vp4 = (const v4f*)vp;
#pragma unroll
      for (int i = 0; i < 8; ++i) {
        v4f t = vp4[i];
#pragma unroll
        for (int j = 0; j < 4; ++j) Vst[wv][i * 4 + j][lane] = (_Float16)t[j];
      }
    }
    // K^T fragments straight from global: lane = key, 16 consecutive dims
    int key0 = k0 + ln, key1 = k0 + 16 + ln;
    const float* kp0 =
        Kb + ((size_t)b * Lk + ((key0 < Lk) ? key0 : (Lk - 1))) * Dm + h * HD + g * 16;
    const float* kp1 =
        Kb + ((size_t)b * Lk + ((key1 < Lk) ? key1 : (Lk - 1))) * Dm + h * HD + g * 16;
    v16h bk0 = ld16h(kp0);
    v16h bk1 = ld16h(kp1);
    v8f z = {};
    v8f s0 = wmma_f16(qa, bk0, z);
    v8f s1 = wmma_f16(qa, bk1, z);
    // bias / masking + online softmax (C layout: row = g*8+r, col = ln)
#pragma unroll
    for (int r = 0; r < 8; ++r) {
      int q = qt * 16 + g * 8 + r;
      float bb0 = 0.f, bb1 = 0.f;
      if (bias_mode == BIAS_TEXT) {
        bb0 = (key0 < Lk && tmask[b * Lk + key0]) ? 0.f : -1e30f;
        bb1 = (key1 < Lk && tmask[b * Lk + key1]) ? 0.f : -1e30f;
      } else if (bias_mode == BIAS_VIS && q >= 1 && q < Lq) {
        int qq = q - 1;
        if (key0 < Lk) {
          int y = key0 / Wd, x = key0 - y * Wd;
          bb0 = dy_e[(((size_t)b * 200 + qq) * Hh + y) * 8 + h] +
                dx_e[(((size_t)b * 200 + qq) * Wd + x) * 8 + h];
        }
        if (key1 < Lk) {
          int y = key1 / Wd, x = key1 - y * Wd;
          bb1 = dy_e[(((size_t)b * 200 + qq) * Hh + y) * 8 + h] +
                dx_e[(((size_t)b * 200 + qq) * Wd + x) * 8 + h];
        }
      }
      if (key0 >= Lk) bb0 = -1e30f;
      if (key1 >= Lk) bb1 = -1e30f;
      float sv0 = s0[r] + bb0, sv1 = s1[r] + bb1;
      float mx = fmaxf(sv0, sv1);
#pragma unroll
      for (int off = 1; off < 16; off <<= 1) mx = fmaxf(mx, __shfl_xor(mx, off, 16));
      float nm = fmaxf(rm[r], mx);
      float sc = __expf(rm[r] - nm);
      float p0 = __expf(sv0 - nm), p1 = __expf(sv1 - nm);
      float rs = p0 + p1;
#pragma unroll
      for (int off = 1; off < 16; off <<= 1) rs += __shfl_xor(rs, off, 16);
      rl[r] = rl[r] * sc + rs;
      rm[r] = nm;
      o0[r] *= sc; o1[r] *= sc;
      Ps[wv][(g * 8 + r) * 32 + ln] = (_Float16)p0;
      Ps[wv][(g * 8 + r) * 32 + 16 + ln] = (_Float16)p1;
    }
    asm volatile("s_wait_dscnt 0x0" ::: "memory");  // wave-local LDS RAW (Ps, Vst)
    // P as A fragment (vector loads), V as B fragments (vector loads)
    const _Float16* pr = &Ps[wv][ln * 32];
    v16h pa = comb(*(const v8h*)(pr + g * 8), *(const v8h*)(pr + 16 + g * 8));
    const _Float16* vr0 = &Vst[wv][ln][0];
    const _Float16* vr1 = &Vst[wv][16 + ln][0];
    v16h bv0 = comb(*(const v8h*)(vr0 + g * 8), *(const v8h*)(vr0 + 16 + g * 8));
    v16h bv1 = comb(*(const v8h*)(vr1 + g * 8), *(const v8h*)(vr1 + 16 + g * 8));
    o0 = wmma_f16(pa, bv0, o0);
    o1 = wmma_f16(pa, bv1, o1);
  }

  // per-wave partials -> LDS, then block merge
  if (ln == 0) {
#pragma unroll
    for (int r = 0; r < 8; ++r) { Ms[wv][g * 8 + r] = rm[r]; Ls[wv][g * 8 + r] = rl[r]; }
  }
#pragma unroll
  for (int r = 0; r < 8; ++r) {
    Os[wv][(g * 8 + r) * 32 + ln] = o0[r];
    Os[wv][(g * 8 + r) * 32 + 16 + ln] = o1[r];
  }
  __syncthreads();
  for (int idx = tid; idx < 16 * 32; idx += 128) {
    int row = idx >> 5, col = idx & 31;
    float mstar = -1e30f;
#pragma unroll
    for (int w = 0; w < 4; ++w) mstar = fmaxf(mstar, Ms[w][row]);
    float lsum = 0.f, osum = 0.f;
#pragma unroll
    for (int w = 0; w < 4; ++w) {
      float e = __expf(Ms[w][row] - mstar);
      lsum += Ls[w][row] * e;
      osum += Os[w][row * 32 + col] * e;
    }
    int q = qt * 16 + row;
    if (q < Lq)
      O[((size_t)b * Lq + q) * Dm + h * HD + col] = osum / fmaxf(lsum, 1e-30f);
  }
}

// ---------------------------------------------------------------------------
// LayerNorm over last dim 256. Row rr -> batch bb=rr/rpb, qi=rr%rpb.
// ---------------------------------------------------------------------------
__global__ __launch_bounds__(256) void k_layernorm(
    const float* __restrict__ X, float* __restrict__ Y,
    const float* __restrict__ gm, const float* __restrict__ bt, int rows,
    int rpb, long ibs, long irs, long obs, long ors) {
  int rr = blockIdx.x;
  if (rr >= rows) return;
  int bb = rr / rpb, qi = rr % rpb;
  const float* x = X + (size_t)bb * ibs + (size_t)qi * irs;
  float v = x[threadIdx.x];
  __shared__ float s1[8], s2[8];
  float a = v, b2 = v * v;
#pragma unroll
  for (int off = 16; off; off >>= 1) {
    a += __shfl_xor(a, off, 32);
    b2 += __shfl_xor(b2, off, 32);
  }
  int wv = threadIdx.x >> 5, ln = threadIdx.x & 31;
  if (!ln) { s1[wv] = a; s2[wv] = b2; }
  __syncthreads();
  float sum = 0.f, sq = 0.f;
#pragma unroll
  for (int i = 0; i < 8; ++i) { sum += s1[i]; sq += s2[i]; }
  float mean = sum * (1.f / 256.f);
  float var = sq * (1.f / 256.f) - mean * mean;
  Y[(size_t)bb * obs + (size_t)qi * ors + threadIdx.x] =
      (v - mean) * rsqrtf(var + 1e-5f) * gm[threadIdx.x] + bt[threadIdx.x];
}

// ---------------------------------------------------------------------------
// Small elementwise / helper kernels
// ---------------------------------------------------------------------------
__global__ void k_add(const float* a, const float* b, float* c, int n) {
  int i = blockIdx.x * blockDim.x + threadIdx.x;
  if (i < n) c[i] = a[i] + b[i];
}

__global__ void k_init_hs(const float* pt, const float* qe, float* hs, int n) {
  int i = blockIdx.x * blockDim.x + threadIdx.x;
  if (i >= n) return;
  int c = i & 255, row = (i >> 8) % 201;
  hs[i] = (row == 0) ? pt[c] : qe[(row - 1) * 256 + c];
}

__global__ void k_init_ref(const float* rp, float* ref, float* out0, int n) {
  int i = blockIdx.x * blockDim.x + threadIdx.x;
  if (i >= n) return;
  int per = i % 800;  // 200*4
  float v = 1.f / (1.f + __expf(-rp[per]));
  ref[i] = v;
  out0[i] = v;
}

__global__ void k_pad_qpos(const float* qp, float* qpp, int n) {
  int i = blockIdx.x * blockDim.x + threadIdx.x;
  if (i >= n) return;
  int c = i & 255, row = (i >> 8) % 201, b = i / (201 * 256);
  qpp[i] = (row == 0) ? 0.f : qp[(((size_t)b * 200) + row - 1) * 256 + c];
}

// sine embedding: (B,200,4 boxes) -> (B,200,512); order [cy,cx,w,h]
__global__ void k_sine(const float* ref, float* out, int n) {
  int i = blockIdx.x * blockDim.x + threadIdx.x;
  if (i >= n) return;
  int c = i & 511, row = i >> 9;
  int comp = c >> 7, j = c & 127, k = j >> 1;
  const float* bx = ref + (size_t)row * 4;
  float x = (comp == 0) ? bx[1] : (comp == 1) ? bx[0] : (comp == 2) ? bx[2] : bx[3];
  float t = __powf(10000.f, (2.f * (float)k) * (1.f / 128.f));
  float pos = x * 6.283185307179586f / t;
  out[i] = (j & 1) ? __cosf(pos) : __sinf(pos);
}

// warp(d) features for rpb MLPs: out (B,200,80,2)
__global__ void k_warp(const float* ref, float* out, int n, int axis, float inv) {
  int i = blockIdx.x * blockDim.x + threadIdx.x;
  if (i >= n) return;
  int s = i & 1;
  int yy = (i >> 1) % 80;
  int row = (i >> 1) / 80;
  const float* bx = ref + (size_t)row * 4;
  float c = axis ? bx[0] : bx[1];
  float sz = axis ? bx[2] : bx[3];
  float edge = c + (s ? 0.5f * sz : -0.5f * sz);
  float d8 = ((float)yy * inv - edge) * 8.f;
  out[i] = copysignf(__log2f(fabsf(d8) + 1.f) * (1.f / 3.f), d8);
}

// fused tiny MLP: 2 -> 256 (relu) -> 8 ; one block per row
__global__ __launch_bounds__(256) void k_rpb_mlp(
    const float* __restrict__ X, const float* __restrict__ W0,
    const float* __restrict__ b0, const float* __restrict__ W1,
    const float* __restrict__ b1, float* __restrict__ Out, int rows) {
  int row = blockIdx.x;
  if (row >= rows) return;
  int t = threadIdx.x;
  float x0 = X[(size_t)row * 2], x1 = X[(size_t)row * 2 + 1];
  float hval = fmaxf(x0 * W0[t] + x1 * W0[256 + t] + b0[t], 0.f);
  const v4f* w1 = (const v4f*)(W1 + t * 8);
  v4f wa = w1[0], wb = w1[1];
  float acc[8];
#pragma unroll
  for (int j = 0; j < 4; ++j) { acc[j] = hval * wa[j]; acc[4 + j] = hval * wb[j]; }
#pragma unroll
  for (int off = 16; off; off >>= 1)
#pragma unroll
    for (int j = 0; j < 8; ++j) acc[j] += __shfl_xor(acc[j], off, 32);
  __shared__ float part[8][8];
  int wv = t >> 5, ln = t & 31;
  if (!ln)
#pragma unroll
    for (int j = 0; j < 8; ++j) part[wv][j] = acc[j];
  __syncthreads();
  if (t < 8) {
    float s = b1[t];
#pragma unroll
    for (int w = 0; w < 8; ++w) s += part[w][t];
    Out[(size_t)row * 8 + t] = s;
  }
}

__global__ void k_box_update(const float* raw, float* ref, float* out, int n) {
  int i = blockIdx.x * blockDim.x + threadIdx.x;
  if (i >= n) return;
  float x = fminf(fmaxf(ref[i], 0.f), 1.f);
  float inv = __logf(fmaxf(x, 1e-3f) / fmaxf(1.f - x, 1e-3f));
  float nr = 1.f / (1.f + __expf(-(raw[i] + inv)));
  ref[i] = nr;
  if (out) out[i] = nr;
}

__global__ void k_pres_out(const float* pv, float* out, int n) {
  int i = blockIdx.x * blockDim.x + threadIdx.x;
  if (i < n) out[i] = fminf(fmaxf(pv[i], -10.f), 10.f);
}

// ---------------------------------------------------------------------------
// Host orchestration
// ---------------------------------------------------------------------------
static inline void gemm(hipStream_t s, const float* A, const float* W,
                        const float* bias, const float* resid, float* C, int M,
                        int N, int K, int flags) {
  dim3 g((M + 127) / 128, (N + 63) / 64);
  k_gemm<<<g, 256, 0, s>>>(A, W, bias, resid, C, M, N, K, flags);
}

extern "C" void kernel_launch(void* const* d_in, const int* in_sizes, int n_in,
                              void* d_out, int out_size, void* d_ws,
                              size_t ws_size, hipStream_t stream) {
  (void)in_sizes; (void)n_in; (void)out_size; (void)ws_size;
  const int B = 2, NQ = 200, D = 256, NH = 8, FF = 2048;
  const int Lq = 201, HW = 6400, Hh = 80, Wd = 80, K = 32;
  const int MQ = B * Lq;          // 402
  const int MV = B * HW;          // 12800

  const float* vf = (const float*)d_in[0];
  const float* tf = (const float*)d_in[1];
  const float* vpe = (const float*)d_in[2];
  const bool* tmask = (const bool*)d_in[3];
  auto F = [&](int i) -> const float* { return (const float*)d_in[i]; };

  float* out = (float*)d_out;
  const size_t inters_off = 0;       // (6,2,200,256)
  const size_t iboxes_off = 614400;  // (6,2,200,4)
  const size_t ipres_off = 624000;   // (6,2,1)

  float* w = (float*)d_ws;
  auto alloc = [&](size_t n) { float* p = w; w += n; return p; };
  float* hs     = alloc((size_t)MQ * D);
  float* qpos_p = alloc((size_t)MQ * D);
  float* qbuf   = alloc((size_t)MQ * D);
  float* Qp     = alloc((size_t)MQ * D);
  float* attn   = alloc((size_t)MQ * D);
  float* tmp_hs = alloc((size_t)MQ * D);
  float* kp_in  = alloc((size_t)MV * D);
  float* Kp     = alloc((size_t)MV * D);
  float* Vp     = alloc((size_t)MV * D);
  float* ffn_h  = alloc((size_t)MQ * FF);
  float* sine   = alloc((size_t)B * NQ * 512);
  float* rph    = alloc((size_t)B * NQ * D);
  float* qpos   = alloc((size_t)B * NQ * D);
  float* dyw    = alloc((size_t)B * NQ * Hh * 2);
  float* dxw    = alloc((size_t)B * NQ * Wd * 2);
  float* dy_e   = alloc((size_t)B * NQ * Hh * 8);
  float* dx_e   = alloc((size_t)B * NQ * Wd * 8);
  float* ref    = alloc((size_t)B * NQ * 4);
  float* bt1    = alloc((size_t)B * NQ * D);
  float* bt2    = alloc((size_t)B * NQ * D);
  float* bo4    = alloc((size_t)B * NQ * 4);
  float* ph     = alloc((size_t)B * D);
  float* pt1    = alloc((size_t)B * D);
  float* pt2    = alloc((size_t)B * D);
  float* pv     = alloc((size_t)B);

  auto elw = [&](int n) { return dim3((n + 255) / 256); };

  // init
  k_init_hs<<<elw(MQ * D), 256, 0, stream>>>(F(238), F(239), hs, MQ * D);
  k_init_ref<<<elw(B * NQ * 4), 256, 0, stream>>>(F(244), ref, out + iboxes_off,
                                                  B * NQ * 4);
  k_add<<<elw(MV * D), 256, 0, stream>>>(vf, vpe, kp_in, MV * D);

  const int QT = (Lq + 15) / 16;  // 13 query tiles
  for (int l = 0; l < 6; ++l) {
    const int lb = 12 + l * 36;
    // leaf map (alphabetical pytree order):
    // mlp: +0 W0 +1 b0 +2 W1 +3 b1 | mlp_ln: +4 g +5 b
    // sa:  +6 bk +7 bo +8 bq +9 bv +10 wk +11 wo +12 wq +13 wv | sa_ln +14,+15
    // ta:  +16..+23 | ta_ln +24,+25 | va: +26..+33 | va_ln +34,+35

    // qpos = MLP(ref_point_head, sine_embed(ref))
    k_sine<<<elw(B * NQ * 512), 256, 0, stream>>>(ref, sine, B * NQ * 512);
    gemm(stream, sine, F(240), F(241), nullptr, rph, B * NQ, D, 512, GF_RELU);
    gemm(stream, rph, F(242), F(243), nullptr, qpos, B * NQ, D, D, 0);
    k_pad_qpos<<<elw(MQ * D), 256, 0, stream>>>(qpos, qpos_p, MQ * D);

    // relative position bias components
    k_warp<<<elw(B * NQ * Hh * 2), 256, 0, stream>>>(ref, dyw, B * NQ * Hh * 2,
                                                     0, 1.f / Hh);
    k_warp<<<elw(B * NQ * Wd * 2), 256, 0, stream>>>(ref, dxw, B * NQ * Wd * 2,
                                                     1, 1.f / Wd);
    k_rpb_mlp<<<B * NQ * Hh, 256, 0, stream>>>(dyw, F(249), F(250), F(251),
                                               F(252), dy_e, B * NQ * Hh);
    k_rpb_mlp<<<B * NQ * Wd, 256, 0, stream>>>(dxw, F(245), F(246), F(247),
                                               F(248), dx_e, B * NQ * Wd);

    // ---- self attention ----
    k_add<<<elw(MQ * D), 256, 0, stream>>>(hs, qpos_p, qbuf, MQ * D);
    gemm(stream, qbuf, F(lb + 12), F(lb + 8), nullptr, Qp, MQ, D, D, 0);
    gemm(stream, qbuf, F(lb + 10), F(lb + 6), nullptr, Kp, MQ, D, D, 0);
    gemm(stream, hs,   F(lb + 13), F(lb + 9), nullptr, Vp, MQ, D, D, 0);
    k_flash<<<dim3(QT, NH, B), 128, 0, stream>>>(Qp, Kp, Vp, attn, nullptr,
                                                 nullptr, nullptr, Lq, Lq, Hh,
                                                 Wd, BIAS_NONE);
    gemm(stream, attn, F(lb + 11), F(lb + 7), hs, tmp_hs, MQ, D, D, GF_RESID);
    k_layernorm<<<MQ, 256, 0, stream>>>(tmp_hs, hs, F(lb + 14), F(lb + 15), MQ,
                                        Lq, Lq * D, D, Lq * D, D);

    // ---- text cross attention ----
    k_add<<<elw(MQ * D), 256, 0, stream>>>(hs, qpos_p, qbuf, MQ * D);
    gemm(stream, qbuf, F(lb + 22), F(lb + 18), nullptr, Qp, MQ, D, D, 0);
    gemm(stream, tf,   F(lb + 20), F(lb + 16), nullptr, Kp, B * K, D, D, 0);
    gemm(stream, tf,   F(lb + 23), F(lb + 19), nullptr, Vp, B * K, D, D, 0);
    k_flash<<<dim3(QT, NH, B), 128, 0, stream>>>(Qp, Kp, Vp, attn, nullptr,
                                                 nullptr, tmask, Lq, K, Hh, Wd,
                                                 BIAS_TEXT);
    gemm(stream, attn, F(lb + 21), F(lb + 17), hs, tmp_hs, MQ, D, D, GF_RESID);
    k_layernorm<<<MQ, 256, 0, stream>>>(tmp_hs, hs, F(lb + 24), F(lb + 25), MQ,
                                        Lq, Lq * D, D, Lq * D, D);

    // ---- vision cross attention ----
    k_add<<<elw(MQ * D), 256, 0, stream>>>(hs, qpos_p, qbuf, MQ * D);
    gemm(stream, qbuf,  F(lb + 32), F(lb + 28), nullptr, Qp, MQ, D, D, 0);
    gemm(stream, kp_in, F(lb + 30), F(lb + 26), nullptr, Kp, MV, D, D, 0);
    gemm(stream, vf,    F(lb + 33), F(lb + 29), nullptr, Vp, MV, D, D, 0);
    k_flash<<<dim3(QT, NH, B), 128, 0, stream>>>(Qp, Kp, Vp, attn, dy_e, dx_e,
                                                 nullptr, Lq, HW, Hh, Wd,
                                                 BIAS_VIS);
    gemm(stream, attn, F(lb + 31), F(lb + 27), hs, tmp_hs, MQ, D, D, GF_RESID);
    k_layernorm<<<MQ, 256, 0, stream>>>(tmp_hs, hs, F(lb + 34), F(lb + 35), MQ,
                                        Lq, Lq * D, D, Lq * D, D);

    // ---- FFN ----
    gemm(stream, hs, F(lb + 0), F(lb + 1), nullptr, ffn_h, MQ, FF, D, GF_RELU);
    gemm(stream, ffn_h, F(lb + 2), F(lb + 3), hs, tmp_hs, MQ, D, FF, GF_RESID);
    k_layernorm<<<MQ, 256, 0, stream>>>(tmp_hs, hs, F(lb + 4), F(lb + 5), MQ,
                                        Lq, Lq * D, D, Lq * D, D);

    // ---- heads ----
    float* qn = out + inters_off + (size_t)l * B * NQ * D;
    k_layernorm<<<B * NQ, 256, 0, stream>>>(hs + D, qn, F(228), F(229), B * NQ,
                                            NQ, Lq * D, D, NQ * D, D);
    gemm(stream, qn, F(6), F(7), nullptr, bt1, B * NQ, D, D, GF_RELU);
    gemm(stream, bt1, F(8), F(9), nullptr, bt2, B * NQ, D, D, GF_RELU);
    gemm(stream, bt2, F(10), F(11), nullptr, bo4, B * NQ, 4, D, 0);
    float* boxout = (l < 5) ? (out + iboxes_off + (size_t)(l + 1) * B * NQ * 4)
                            : nullptr;
    k_box_update<<<elw(B * NQ * 4), 256, 0, stream>>>(bo4, ref, boxout,
                                                      B * NQ * 4);
    k_layernorm<<<B, 256, 0, stream>>>(hs, ph, F(236), F(237), B, 1, Lq * D, D,
                                       D, D);
    gemm(stream, ph, F(230), F(231), nullptr, pt1, B, D, D, GF_RELU);
    gemm(stream, pt1, F(232), F(233), nullptr, pt2, B, D, D, GF_RELU);
    gemm(stream, pt2, F(234), F(235), nullptr, pv, B, 1, D, 0);
    k_pres_out<<<1, 32, 0, stream>>>(pv, out + ipres_off + (size_t)l * B, B);
  }
}
